// TransformerEncoder_56521769615863
// MI455X (gfx1250) — matrix operand order repaired and checked by reading, not compiled, not run on hardware
//
#include <hip/hip_runtime.h>
#include <hip/hip_bf16.h>

// ---------------------------------------------------------------------------
// Transformer encoder block for MI455X (gfx1250, wave32, WMMA bf16 16x16x32)
// B=8 S=1024 D=768 H=12 HD=64 FF=3072.  All GEMMs via v_wmma_f32_16x16x32_bf16.
// Attention computed transposed (S^T = K Q^T, O^T = V^T P^T) so each lane owns
// one softmax row: reductions need a single xor16 shuffle and packed LDS I/O.
// ---------------------------------------------------------------------------

typedef __bf16 bf16_t;
typedef __attribute__((ext_vector_type(16))) __bf16 v16bf;
typedef __attribute__((ext_vector_type(8)))  __bf16 v8bf;
typedef __attribute__((ext_vector_type(8)))  float  v8f;

#define DIM   768
#define SEQ   1024
#define NH    12
#define HDIM  64
#define FFD   3072
#define NTOK  8192   // B*S

__device__ inline v8f wmma_bf16(v16bf a, v16bf b, v8f c) {
  return __builtin_amdgcn_wmma_f32_16x16x32_bf16(false, a, false, b, (short)0, c,
                                                 false, false);
}

// A fragment: 16x32 bf16, contiguous K order (same as B): lane(0-15)=row M, half=lane>>4.
// elems 0..7  -> K = k0 + 16*half + e      (16B contiguous)
// elems 8..15 -> K = k0 + 16*half + 8 + e  (16B contiguous)
__device__ inline v16bf load_a_frag(const bf16_t* __restrict__ base, int lda,
                                    int row, int k0, int lane) {
  int half = lane >> 4, lm = lane & 15;
  const bf16_t* p0 = base + (size_t)(row + lm) * lda + k0 + 16 * half;
  v8bf lo = *(const v8bf*)p0;
  v8bf hi = *(const v8bf*)(p0 + 8);
  v16bf a;
#pragma unroll
  for (int i = 0; i < 8; ++i) { a[i] = lo[i]; a[i + 8] = hi[i]; }
  return a;
}

// B fragment: 32x16 bf16 from Bt[N][K] (K contiguous per column):
// lane col N = col0 + lm, elems e -> K = k0 + 16*half + e  (one 32B load)
__device__ inline v16bf load_b_frag(const bf16_t* __restrict__ bt, int ldb,
                                    int col0, int k0, int lane) {
  int half = lane >> 4, lm = lane & 15;
  return *(const v16bf*)(bt + (size_t)(col0 + lm) * ldb + k0 + 16 * half);
}

__device__ inline float gelu_f(float v) {
  return 0.5f * v * (1.0f + erff(v * 0.70710678118654752f));
}

// ---------------------------------------------------------------------------
// LayerNorm + cast to bf16.  One block per token row (768 = 256*3).
// ---------------------------------------------------------------------------
__global__ __launch_bounds__(256) void ln_cast_kernel(
    const float* __restrict__ x, const float* __restrict__ w,
    const float* __restrict__ b, bf16_t* __restrict__ y) {
  __shared__ float red[256];
  int row = blockIdx.x, t = threadIdx.x;
  const float* xr = x + (size_t)row * DIM;
  float v0 = xr[t], v1 = xr[t + 256], v2 = xr[t + 512];
  red[t] = v0 + v1 + v2;
  __syncthreads();
#pragma unroll
  for (int o = 128; o > 0; o >>= 1) {
    if (t < o) red[t] += red[t + o];
    __syncthreads();
  }
  float mu = red[0] * (1.0f / DIM);
  __syncthreads();
  float d0 = v0 - mu, d1 = v1 - mu, d2 = v2 - mu;
  red[t] = d0 * d0 + d1 * d1 + d2 * d2;
  __syncthreads();
#pragma unroll
  for (int o = 128; o > 0; o >>= 1) {
    if (t < o) red[t] += red[t + o];
    __syncthreads();
  }
  float rstd = rsqrtf(red[0] * (1.0f / DIM) + 1e-5f);
  bf16_t* yr = y + (size_t)row * DIM;
  yr[t]       = (bf16_t)(d0 * rstd * w[t]       + b[t]);
  yr[t + 256] = (bf16_t)(d1 * rstd * w[t + 256] + b[t + 256]);
  yr[t + 512] = (bf16_t)(d2 * rstd * w[t + 512] + b[t + 512]);
}

// ---------------------------------------------------------------------------
// One-time weight repacks (fp32 -> bf16, K-contiguous per output column).
// ---------------------------------------------------------------------------
// Wq/Wk/Wv [H][D][HD] -> Wt[n][k], n = h*64+e, k = d
__global__ __launch_bounds__(256) void cast_qkv_w(const float* __restrict__ W,
                                                  bf16_t* __restrict__ Wt) {
  int idx = blockIdx.x * 256 + threadIdx.x;  // over 768*768
  int n = idx / DIM, k = idx - n * DIM;
  int h = n >> 6, e = n & 63;
  Wt[idx] = (bf16_t)W[((size_t)h * DIM + k) * HDIM + e];
}
// Generic: src[K][N] fp32 -> dst[N][K] bf16
__global__ __launch_bounds__(256) void cast_tr(const float* __restrict__ W,
                                               bf16_t* __restrict__ Wt,
                                               int Kd, int Nd) {
  int idx = blockIdx.x * 256 + threadIdx.x;  // over Nd*Kd
  int n = idx / Kd, k = idx - n * Kd;
  Wt[idx] = (bf16_t)W[(size_t)k * Nd + n];
}

// ---------------------------------------------------------------------------
// WMMA GEMM: C[M=8192, N] = A[M,K] * Bt[N,K]^T, mode-specific epilogue/store.
// Block = 256 thr = 8 waves (4M x 2N), block tile 128x64, wave tile 32x32.
// ---------------------------------------------------------------------------
enum { M_QK = 0, M_V = 1, M_FF1 = 2, M_PROJ = 3, M_FF2 = 4 };

template <int MODE>
__global__ __launch_bounds__(256) void gemm_kernel(
    const bf16_t* __restrict__ A, const bf16_t* __restrict__ Bt,
    const float* __restrict__ bias, const float* __restrict__ resid,
    void* __restrict__ outp, int K) {
  int lane = threadIdx.x & 31, w = threadIdx.x >> 5;
  int half = lane >> 4, lm = lane & 15;
  int wm = w & 3, wn = w >> 2;
  int row0 = blockIdx.y * 128 + wm * 32;
  int col0 = blockIdx.x * 64 + wn * 32;
  const bf16_t* Arow = A + (size_t)row0 * K;

  v8f acc[2][2] = {};
  for (int k0 = 0; k0 < K; k0 += 32) {
    v16bf a0 = load_a_frag(Arow, K, 0, k0, lane);
    v16bf a1 = load_a_frag(Arow, K, 16, k0, lane);
    v16bf b0 = load_b_frag(Bt, K, col0, k0, lane);
    v16bf b1 = load_b_frag(Bt, K, col0 + 16, k0, lane);
    acc[0][0] = wmma_bf16(a0, b0, acc[0][0]);
    acc[0][1] = wmma_bf16(a0, b1, acc[0][1]);
    acc[1][0] = wmma_bf16(a1, b0, acc[1][0]);
    acc[1][1] = wmma_bf16(a1, b1, acc[1][1]);
  }

#pragma unroll
  for (int mi = 0; mi < 2; ++mi)
#pragma unroll
    for (int ni = 0; ni < 2; ++ni) {
      v8f c = acc[mi][ni];
#pragma unroll
      for (int r = 0; r < 8; ++r) {
        int row = row0 + mi * 16 + 8 * half + r;  // token index (b*S+s)
        int col = col0 + ni * 16 + lm;            // output feature
        float v = c[r] + bias[col];
        if (MODE == M_QK) {
          // store [b][h][s][d] bf16
          int h = col >> 6, e = col & 63;
          int b = row >> 10, s = row & 1023;
          ((bf16_t*)outp)[((((size_t)b * NH + h) * SEQ) + s) * HDIM + e] =
              (bf16_t)v;
        } else if (MODE == M_V) {
          // store transposed [b][h][d][s] bf16 for attention A-fragments
          int h = col >> 6, e = col & 63;
          int b = row >> 10, s = row & 1023;
          ((bf16_t*)outp)[(((size_t)b * NH + h) * HDIM + e) * SEQ + s] =
              (bf16_t)v;
        } else if (MODE == M_FF1) {
          ((bf16_t*)outp)[(size_t)row * FFD + col] = (bf16_t)gelu_f(v);
        } else if (MODE == M_PROJ) {
          ((float*)outp)[(size_t)row * DIM + col] =
              gelu_f(v) + resid[(size_t)row * DIM + col];
        } else {  // M_FF2
          ((float*)outp)[(size_t)row * DIM + col] =
              gelu_f(v) + resid[(size_t)row * DIM + col];
        }
      }
    }
}

// ---------------------------------------------------------------------------
// Flash attention, transposed (no 1/sqrt(d) scale, per reference).
// Block = 4 waves; each wave owns 16 query columns.  Per 32-key step:
//   S^T tiles via wmma (A = K tile, B = Q^T), online softmax with one xor16
//   shuffle per reduction (lane == query), P^T packed to LDS as [q][key]
//   with ds_store_b128, O^T = V^T P^T accumulated via wmma, V^T A-frags
//   straight from the transposed Vt produced by the V projection.
// ---------------------------------------------------------------------------
__global__ __launch_bounds__(128) void attn_kernel(
    const bf16_t* __restrict__ Qh, const bf16_t* __restrict__ Kh,
    const bf16_t* __restrict__ Vt, bf16_t* __restrict__ O) {
  __shared__ bf16_t pls[4][16 * 32];  // per-wave P^T staging: [query][key0..31]
  int lane = threadIdx.x & 31, w = threadIdx.x >> 5;
  int half = lane >> 4, lm = lane & 15;
  int bh = blockIdx.y;
  int b = bh / NH, h = bh - b * NH;
  int q0 = blockIdx.x * 64 + w * 16;

  const bf16_t* Qb = Qh + ((size_t)bh * SEQ + q0) * HDIM;
  const bf16_t* Kb = Kh + (size_t)bh * SEQ * HDIM;
  const bf16_t* Vb = Vt + (size_t)bh * HDIM * SEQ;

  // Q^T as B fragments (col = query, K contiguous): loaded once
  v16bf bq0 = load_b_frag(Qb, HDIM, 0, 0, lane);   // d = 0..31
  v16bf bq1 = load_b_frag(Qb, HDIM, 0, 32, lane);  // d = 32..63

  // O^T accumulators: 4 frags of (16 dims x 16 queries); lane col = query lm
  v8f o0 = {}, o1 = {}, o2 = {}, o3 = {};
  float m = -1e30f, l = 0.0f;  // per-lane = per-query stats

  for (int kt = 0; kt < SEQ; kt += 32) {
    // S^T tiles: rows = keys, cols = queries
    v8f s0 = {}, s1 = {};
    s0 = wmma_bf16(load_a_frag(Kb, HDIM, kt, 0, lane), bq0, s0);
    s0 = wmma_bf16(load_a_frag(Kb, HDIM, kt, 32, lane), bq1, s0);
    s1 = wmma_bf16(load_a_frag(Kb, HDIM, kt + 16, 0, lane), bq0, s1);
    s1 = wmma_bf16(load_a_frag(Kb, HDIM, kt + 16, 32, lane), bq1, s1);

    // tile max over this lane's 16 keys, then across the two halves
    float tm = s0[0];
#pragma unroll
    for (int r = 1; r < 8; ++r) tm = fmaxf(tm, s0[r]);
#pragma unroll
    for (int r = 0; r < 8; ++r) tm = fmaxf(tm, s1[r]);
    tm = fmaxf(tm, __shfl_xor(tm, 16, 32));

    float nm = fmaxf(m, tm);
    float corr = __expf(m - nm);
    m = nm;

    float rs = 0.0f;
    v8bf p0, p1;
#pragma unroll
    for (int r = 0; r < 8; ++r) {
      float e0 = __expf(s0[r] - nm);
      float e1 = __expf(s1[r] - nm);
      p0[r] = (bf16_t)e0;
      p1[r] = (bf16_t)e1;
      rs += e0 + e1;
    }
    rs += __shfl_xor(rs, 16, 32);
    l = l * corr + rs;

    // rescale O^T (uniform per lane)
#pragma unroll
    for (int r = 0; r < 8; ++r) {
      o0[r] *= corr; o1[r] *= corr; o2[r] *= corr; o3[r] *= corr;
    }

    // P^T -> LDS as [query][key] (8 contiguous keys per half per tile)
    *(v8bf*)&pls[w][lm * 32 + 8 * half]      = p0;  // keys kt    + 8h..+7
    *(v8bf*)&pls[w][lm * 32 + 16 + 8 * half] = p1;  // keys kt+16 + 8h..+7
    v16bf pb = load_b_frag(pls[w], 32, 0, 0, lane);

    // O^T += V^T(16 dims x 32 keys) * P^T(32 keys x 16 queries)
    o0 = wmma_bf16(load_a_frag(Vb, SEQ, 0, kt, lane), pb, o0);
    o1 = wmma_bf16(load_a_frag(Vb, SEQ, 16, kt, lane), pb, o1);
    o2 = wmma_bf16(load_a_frag(Vb, SEQ, 32, kt, lane), pb, o2);
    o3 = wmma_bf16(load_a_frag(Vb, SEQ, 48, kt, lane), pb, o3);
  }

  // normalize and store: lane's query row, 8 contiguous dims per frag-half
  float inv = 1.0f / l;
  int q = q0 + lm;
  bf16_t* orow = O + ((size_t)b * SEQ + q) * DIM + h * HDIM + 8 * half;
  v8bf t0, t1, t2, t3;
#pragma unroll
  for (int r = 0; r < 8; ++r) {
    t0[r] = (bf16_t)(o0[r] * inv);
    t1[r] = (bf16_t)(o1[r] * inv);
    t2[r] = (bf16_t)(o2[r] * inv);
    t3[r] = (bf16_t)(o3[r] * inv);
  }
  *(v8bf*)(orow)      = t0;
  *(v8bf*)(orow + 16) = t1;
  *(v8bf*)(orow + 32) = t2;
  *(v8bf*)(orow + 48) = t3;
}

// ---------------------------------------------------------------------------
// Orchestration
// ---------------------------------------------------------------------------
extern "C" void kernel_launch(void* const* d_in, const int* in_sizes, int n_in,
                              void* d_out, int out_size, void* d_ws,
                              size_t ws_size, hipStream_t stream) {
  const float* x    = (const float*)d_in[0];
  const float* ln1w = (const float*)d_in[1];
  const float* ln1b = (const float*)d_in[2];
  const float* Wq   = (const float*)d_in[3];
  const float* bq   = (const float*)d_in[4];
  const float* Wk   = (const float*)d_in[5];
  const float* bk   = (const float*)d_in[6];
  const float* Wv   = (const float*)d_in[7];
  const float* bv   = (const float*)d_in[8];
  const float* Wp   = (const float*)d_in[9];
  const float* bp   = (const float*)d_in[10];
  const float* ln2w = (const float*)d_in[11];
  const float* ln2b = (const float*)d_in[12];
  const float* W1   = (const float*)d_in[13];
  const float* b1   = (const float*)d_in[14];
  const float* W2   = (const float*)d_in[15];
  const float* b2   = (const float*)d_in[16];

  char* p = (char*)d_ws;
  auto take = [&](size_t bytes) {
    char* r = p;
    p += (bytes + 255) & ~size_t(255);
    return r;
  };
  const size_t TOKD = (size_t)NTOK * DIM;             // 6291456
  bf16_t* yb  = (bf16_t*)take(TOKD * 2);              // LN1(x) bf16
  bf16_t* Qh  = (bf16_t*)take(TOKD * 2);              // [b][h][s][d]
  bf16_t* Kh  = (bf16_t*)take(TOKD * 2);
  bf16_t* Vtb = (bf16_t*)take(TOKD * 2);              // [b][h][d][s]
  bf16_t* Ob  = (bf16_t*)take(TOKD * 2);              // attn out, concat heads
  float*  x1  = (float*)take(TOKD * 4);               // residual 1
  bf16_t* x1n = (bf16_t*)take(TOKD * 2);              // LN2(x1) bf16
  bf16_t* h1  = (bf16_t*)take((size_t)NTOK * FFD * 2);
  bf16_t* Wqt = (bf16_t*)take((size_t)DIM * DIM * 2);
  bf16_t* Wkt = (bf16_t*)take((size_t)DIM * DIM * 2);
  bf16_t* Wvt = (bf16_t*)take((size_t)DIM * DIM * 2);
  bf16_t* Wpt = (bf16_t*)take((size_t)DIM * DIM * 2);
  bf16_t* W1t = (bf16_t*)take((size_t)DIM * FFD * 2);
  bf16_t* W2t = (bf16_t*)take((size_t)FFD * DIM * 2);

  // LN1 + weight repacks
  ln_cast_kernel<<<NTOK, 256, 0, stream>>>(x, ln1w, ln1b, yb);
  cast_qkv_w<<<(DIM * DIM) / 256, 256, 0, stream>>>(Wq, Wqt);
  cast_qkv_w<<<(DIM * DIM) / 256, 256, 0, stream>>>(Wk, Wkt);
  cast_qkv_w<<<(DIM * DIM) / 256, 256, 0, stream>>>(Wv, Wvt);
  cast_tr<<<(DIM * DIM) / 256, 256, 0, stream>>>(Wp, Wpt, DIM, DIM);
  cast_tr<<<(DIM * FFD) / 256, 256, 0, stream>>>(W1, W1t, DIM, FFD);
  cast_tr<<<(FFD * DIM) / 256, 256, 0, stream>>>(W2, W2t, FFD, DIM);

  // QKV projections (N=768 each)
  dim3 gQKV(DIM / 64, NTOK / 128);
  gemm_kernel<M_QK><<<gQKV, 256, 0, stream>>>(yb, Wqt, bq, nullptr, Qh, DIM);
  gemm_kernel<M_QK><<<gQKV, 256, 0, stream>>>(yb, Wkt, bk, nullptr, Kh, DIM);
  gemm_kernel<M_V><<<gQKV, 256, 0, stream>>>(yb, Wvt, bv, nullptr, Vtb, DIM);

  // attention: 16 q-tiles of 64 queries per (b,h)
  attn_kernel<<<dim3(SEQ / 64, 8 * NH), 128, 0, stream>>>(Qh, Kh, Vtb, Ob);

  // proj + GELU + residual -> x1
  gemm_kernel<M_PROJ><<<gQKV, 256, 0, stream>>>(Ob, Wpt, bp, x, x1, DIM);

  // LN2
  ln_cast_kernel<<<NTOK, 256, 0, stream>>>(x1, ln2w, ln2b, x1n);

  // MLP
  gemm_kernel<M_FF1><<<dim3(FFD / 64, NTOK / 128), 256, 0, stream>>>(
      x1n, W1t, b1, nullptr, h1, DIM);
  gemm_kernel<M_FF2><<<gQKV, 256, 0, stream>>>(h1, W2t, b2, x1, (float*)d_out,
                                               FFD);
}
